// EquivarLayerTorch_22385369547193
// MI455X (gfx1250) — compile-verified, matrix operand order checked by reading, not run on hardware
//
#include <hip/hip_runtime.h>
#include <cstdint>
#include <cstddef>

// ---------------------------------------------------------------------------
// EquivarLayer fused pipeline for MI455X (gfx1250, wave32, WMMA).
//
// inputs (setup_inputs order):
//   d_in[0] ind_2 : (640000,2) int64
//   d_in[1] p3    : (20000,3,128) f32
//   d_in[2] i1    : (640000,128) f32
//   d_in[3] d3    : (640000,3) f32
//   d_in[4] W_ii  : (128,128) f32
//   d_in[5] b_ii  : (128,) f32
//   d_in[6] W_pp  : (128,128) f32
// output: p3_new (20000*3*128 f32) ++ dotted (20000*128 f32)
//
// Roofline: i1 (328 MB) is the only HBM-compulsory stream; p3, the segment-sum
// accumulator and both weights live in the 192 MB L2. Floor ~ 17 us @23.3TB/s.
// GEMM flops (~23 GFLOP) are noise vs v_wmma_f32_16x16x32_f16 throughput, so
// the design is bandwidth-shaped: stream i1 once, never materialize i1g,
// segment-sum via L2-resident global_atomic_add_f32.
// ---------------------------------------------------------------------------

typedef __attribute__((ext_vector_type(16))) _Float16 v16h;
typedef __attribute__((ext_vector_type(8)))  float    v8f;

static constexpr int kR       = 128;
static constexpr int kAtoms   = 20000;
static constexpr int kPairs   = 640000;
static constexpr size_t kAccFloats = (size_t)kAtoms * 3 * kR;        // 7,680,000
static constexpr size_t kAccBytes  = kAccFloats * sizeof(float);     // 30,720,000
static constexpr int kPackHalves   = 4 * 8 * 32 * 16;                // 16384 per W

// gfx1250 has a hardware tanh transcendental (V_TANH_F32); use it when the
// builtin exists, else fall back to libm (branchy exp-based expansion).
__device__ __forceinline__ float fast_tanh(float x) {
#if __has_builtin(__builtin_amdgcn_tanhf)
  return __builtin_amdgcn_tanhf(x);
#else
  return tanhf(x);
#endif
}

// ---------------------------------------------------------------------------
// Kernel 0: repack W (128x128 f32, row-major [k][n]) into f16 WMMA-B layout.
// B tile (32x16) for (kstep ks, ntile nt): lane L holds col n = nt*16 + L%16,
// half-group h = L/16; halves 0..7 -> k = ks*32 + h*8 + hh,
// halves 8..15 -> k = ks*32 + 16 + h*8 + (hh-8).  Stored per-lane contiguous
// (32 bytes) so each wave fetches its B operand with two b128 loads from L2.
// ---------------------------------------------------------------------------
__global__ void pack_w(const float* __restrict__ Wii, const float* __restrict__ Wpp,
                       _Float16* __restrict__ packII, _Float16* __restrict__ packPP) {
  int tid = blockIdx.x * blockDim.x + threadIdx.x;   // 0 .. 32767
  const float* W   = (tid < kPackHalves) ? Wii    : Wpp;
  _Float16*   dst  = (tid < kPackHalves) ? packII : packPP;
  int idx  = tid & (kPackHalves - 1);
  int g    = idx >> 9;            // (ks*8 + nt), 0..31
  int lane = (idx >> 4) & 31;
  int hh   = idx & 15;
  int ks   = g >> 3, nt = g & 7;
  int n    = nt * 16 + (lane & 15);
  int h    = lane >> 4;
  int k    = ks * 32 + ((hh < 8) ? (h * 8 + hh) : (16 + h * 8 + (hh - 8)));
  dst[idx] = (_Float16)W[k * kR + n];
}

// ---------------------------------------------------------------------------
// Kernel 1: zero the segment-sum accumulator (ws) — required every launch.
// ---------------------------------------------------------------------------
__global__ void zero_acc(float4* __restrict__ p) {
  size_t i = (size_t)blockIdx.x * blockDim.x + threadIdx.x;   // 1,920,000 vec4
  p[i] = make_float4(0.f, 0.f, 0.f, 0.f);
}

// ---------------------------------------------------------------------------
// Load a 16x32 f16 A-operand slab for k-step `ks` from an f32 row-major tile.
// A layout: lane L -> row m = L%16, h = L/16; halves 0..7: K = ks*32+h*8+0..7,
// halves 8..15: K = ks*32+16+h*8+0..7.  Four float4 loads per lane.
// ---------------------------------------------------------------------------
__device__ __forceinline__ v16h load_a16(const float* __restrict__ rowp, int ks, int h) {
  const float* b0 = rowp + ks * 32 + h * 8;
  const float* b1 = rowp + ks * 32 + 16 + h * 8;
  float4 f0 = *(const float4*)(b0);
  float4 f1 = *(const float4*)(b0 + 4);
  float4 f2 = *(const float4*)(b1);
  float4 f3 = *(const float4*)(b1 + 4);
  v16h a;
  a[0]=(_Float16)f0.x;  a[1]=(_Float16)f0.y;  a[2]=(_Float16)f0.z;  a[3]=(_Float16)f0.w;
  a[4]=(_Float16)f1.x;  a[5]=(_Float16)f1.y;  a[6]=(_Float16)f1.z;  a[7]=(_Float16)f1.w;
  a[8]=(_Float16)f2.x;  a[9]=(_Float16)f2.y;  a[10]=(_Float16)f2.z; a[11]=(_Float16)f2.w;
  a[12]=(_Float16)f3.x; a[13]=(_Float16)f3.y; a[14]=(_Float16)f3.z; a[15]=(_Float16)f3.w;
  return a;
}

// ---------------------------------------------------------------------------
// Kernel 2 (fused main): per wave = 16 pairs.
//   i1g(16x128) = tanh(i1_tile @ W_ii + b_ii)   via 32x v_wmma_f32_16x16x32_f16
//   then scatter: acc[i,x,r] += (p3[j,x,r] + d3[p,x]) * i1g[p,r]  (f32 atomics,
//   both p3 and acc are L2-resident; i1 streamed from HBM exactly once).
// C layout: lane L, VGPR v of tile t == element (row v+8*(L/16), col t*16+L%16).
// ---------------------------------------------------------------------------
__global__ void __launch_bounds__(256) pair_kernel(
    const long long* __restrict__ ind2, const float* __restrict__ p3,
    const float* __restrict__ i1, const float* __restrict__ d3,
    const _Float16* __restrict__ wpack, const float* __restrict__ b_ii,
    float* __restrict__ acc) {
  const int lane = threadIdx.x & 31;
  const int wave = threadIdx.x >> 5;
  const int tile = blockIdx.x * 8 + wave;          // 40000 tiles exactly
  const long long pairBase = (long long)tile * 16;
  const int m = lane & 15;
  const int h = lane >> 4;

  // ---- A operand: 16 rows of i1, all 4 k-steps kept in registers ----
  const float* rowp = i1 + (size_t)(pairBase + m) * kR;
  v16h A[4];
#pragma unroll
  for (int ks = 0; ks < 4; ++ks) A[ks] = load_a16(rowp, ks, h);

  // ---- GEMM + bias + tanh ----
  const v16h* bp = (const v16h*)wpack;
  v8f g[8];
#pragma unroll
  for (int nt = 0; nt < 8; ++nt) {
    v8f c = {0.f, 0.f, 0.f, 0.f, 0.f, 0.f, 0.f, 0.f};
#pragma unroll
    for (int ks = 0; ks < 4; ++ks) {
      v16h b = bp[(size_t)(ks * 8 + nt) * 32 + lane];
      c = __builtin_amdgcn_wmma_f32_16x16x32_f16(false, A[ks], false, b,
                                                 (short)0, c, false, false);
    }
    const float bias = b_ii[nt * 16 + m];
#pragma unroll
    for (int v = 0; v < 8; ++v) c[v] = fast_tanh(c[v] + bias);
    g[nt] = c;
  }

  // ---- scatter: per VGPR-row v this lane owns pair (pairBase + v + 8h) ----
#pragma unroll
  for (int v = 0; v < 8; ++v) {
    const long long p = pairBase + v + 8 * h;
    const int ii = (int)ind2[p * 2];
    const int jj = (int)ind2[p * 2 + 1];
    const float dx0 = d3[p * 3 + 0];
    const float dx1 = d3[p * 3 + 1];
    const float dx2 = d3[p * 3 + 2];
    const float* pj = p3 + (size_t)jj * (3 * kR);
    float*       pi = acc + (size_t)ii * (3 * kR);
#pragma unroll
    for (int x = 0; x < 3; ++x) {
      const float dxv = (x == 0) ? dx0 : ((x == 1) ? dx1 : dx2);
#pragma unroll
      for (int t = 0; t < 8; ++t) {
        const int r = t * 16 + m;
        const float val = (pj[x * kR + r] + dxv) * g[t][v];
        unsafeAtomicAdd(pi + x * kR + r, val);   // global_atomic_add_f32, L2 scope
      }
    }
  }
}

// ---------------------------------------------------------------------------
// Kernel 3: p3_new = acc(60000x128) @ W_pp, same WMMA pipeline, store to out.
// ---------------------------------------------------------------------------
__global__ void __launch_bounds__(256) gemm2_kernel(
    const float* __restrict__ acc, const _Float16* __restrict__ wpack,
    float* __restrict__ out) {
  const int lane = threadIdx.x & 31;
  const int wave = threadIdx.x >> 5;
  const int tile = blockIdx.x * 8 + wave;
  if (tile >= 3750) return;                        // wave-uniform guard
  const int m = lane & 15;
  const int h = lane >> 4;
  const size_t rowBase = (size_t)tile * 16;        // 3750*16 = 60000 exact

  const float* rowp = acc + (rowBase + m) * kR;
  v16h A[4];
#pragma unroll
  for (int ks = 0; ks < 4; ++ks) A[ks] = load_a16(rowp, ks, h);

  const v16h* bp = (const v16h*)wpack;
#pragma unroll
  for (int nt = 0; nt < 8; ++nt) {
    v8f c = {0.f, 0.f, 0.f, 0.f, 0.f, 0.f, 0.f, 0.f};
#pragma unroll
    for (int ks = 0; ks < 4; ++ks) {
      v16h b = bp[(size_t)(ks * 8 + nt) * 32 + lane];
      c = __builtin_amdgcn_wmma_f32_16x16x32_f16(false, A[ks], false, b,
                                                 (short)0, c, false, false);
    }
#pragma unroll
    for (int v = 0; v < 8; ++v)
      out[(rowBase + v + 8 * h) * kR + nt * 16 + m] = c[v];
  }
}

// ---------------------------------------------------------------------------
// Kernel 4: dotted[i,r] = sum_x p3_new[i,x,r]^2
// ---------------------------------------------------------------------------
__global__ void dotted_kernel(const float* __restrict__ p3n, float* __restrict__ dotted) {
  const size_t tid = (size_t)blockIdx.x * blockDim.x + threadIdx.x;  // 2,560,000
  const size_t i = tid / kR, r = tid % kR;
  const float* b = p3n + i * (3 * kR) + r;
  const float a0 = b[0], a1 = b[kR], a2 = b[2 * kR];
  dotted[tid] = a0 * a0 + a1 * a1 + a2 * a2;
}

// ---------------------------------------------------------------------------
extern "C" void kernel_launch(void* const* d_in, const int* in_sizes, int n_in,
                              void* d_out, int out_size, void* d_ws, size_t ws_size,
                              hipStream_t stream) {
  (void)in_sizes; (void)n_in; (void)out_size; (void)ws_size;
  const long long* ind2 = (const long long*)d_in[0];
  const float* p3  = (const float*)d_in[1];
  const float* i1  = (const float*)d_in[2];
  const float* d3  = (const float*)d_in[3];
  const float* Wii = (const float*)d_in[4];
  const float* bii = (const float*)d_in[5];
  const float* Wpp = (const float*)d_in[6];
  float* out = (float*)d_out;

  char* ws = (char*)d_ws;
  float*    acc    = (float*)ws;                          // 30,720,000 B
  _Float16* packII = (_Float16*)(ws + kAccBytes);         // 32 KB
  _Float16* packPP = packII + kPackHalves;                // 32 KB

  pack_w     <<<128,  256, 0, stream>>>(Wii, Wpp, packII, packPP);
  zero_acc   <<<7500, 256, 0, stream>>>((float4*)acc);    // 1,920,000 float4
  pair_kernel<<<5000, 256, 0, stream>>>(ind2, p3, i1, d3, packII, bii, acc);
  gemm2_kernel<<<469, 256, 0, stream>>>(acc, packPP, out);
  dotted_kernel<<<10000, 256, 0, stream>>>(out, out + kAccFloats);
}
